// MambaModel_70781061038355
// MI455X (gfx1250) — matrix operand order repaired and checked by reading, not compiled, not run on hardware
//
#include <hip/hip_runtime.h>
#include <hip/hip_bf16.h>

// ---------------- model constants (from reference) ----------------
#define NTOKEN   60000
#define D_MODEL  512
#define NLAYERS  4
#define D_INNER  1024
#define D_STATE  16
#define D_CONV   4
#define DT_RANK  32
#define BATCH    8
#define SEQ      2048
#define MTOT     (BATCH * SEQ)       // 16384 rows
#define EPSLN    1e-5f

typedef _Float16 half_t;
typedef __attribute__((ext_vector_type(16))) _Float16 v16h;
typedef __attribute__((ext_vector_type(8)))  _Float16 v8h;
typedef __attribute__((ext_vector_type(8)))  float    v8f;
typedef __attribute__((ext_vector_type(4)))  float    v4f;

__device__ __forceinline__ v16h combine16(v8h lo, v8h hi) {
    v16h r;
#pragma unroll
    for (int j = 0; j < 8; ++j) { r[j] = lo[j]; r[8 + j] = hi[j]; }
    return r;
}

__device__ __forceinline__ float softplusf(float x) {
    if (x > 20.f) return x;
    return log1pf(__expf(x));
}

__device__ __forceinline__ float siluf(float x) {
    return x / (1.f + __expf(-x));
}

// =============================================================
// WMMA GEMM:  C[M x N] = A[M x K] * Wt[K x N]   (A,Wt f16; C f32)
// EPI==0: plain store.  EPI==1: C = softplus(acc + bias[n]).
// Block = 256 threads (8 waves), arranged 4(M-groups) x 2(N-groups).
// Per-wave tile: 32(M) x 16*NT(N)  ->  block tile 128 x 32*NT.
// NT=4: 8 WMMA per 6 fragment loads; NT=2 for narrow N (=64).
// =============================================================
template <int EPI, int NT>
__global__ __launch_bounds__(256) void gemm_wmma_kernel(
    const half_t* __restrict__ A,
    const half_t* __restrict__ Wt,
    float* __restrict__ C,
    const float* __restrict__ bias,
    int M, int N, int K)
{
    const int lane = threadIdx.x & 31;
    const int wave = threadIdx.x >> 5;     // 0..7
    const int g    = lane >> 4;            // A half-group
    const int m16  = lane & 15;

    const int mBlock = blockIdx.y * 128;
    const int nBlock = blockIdx.x * (32 * NT);
    const int wm = (wave & 3) * 32;            // wave M offset inside block
    const int wn = (wave >> 2) * (16 * NT);    // wave N offset inside block

    v8f acc[2][NT];
#pragma unroll
    for (int i = 0; i < 2; ++i)
#pragma unroll
        for (int j = 0; j < NT; ++j) { v8f z = {}; acc[i][j] = z; }

    for (int k0 = 0; k0 < K; k0 += 32) {
        // ---- A fragments (2 M-tiles). 16-bit A layout: lane m holds
        // K = {8g..8g+7} and {16+8g..16+8g+7} -> two 16B contiguous loads.
        v16h afrag[2];
#pragma unroll
        for (int i = 0; i < 2; ++i) {
            const half_t* ap = A + (size_t)(mBlock + wm + i * 16 + m16) * K + k0 + g * 8;
            v8h lo = *(const v8h*)ap;
            v8h hi = *(const v8h*)(ap + 16);
            afrag[i] = combine16(lo, hi);
        }
        // ---- B fragments (NT N-tiles). Lane n = lane&15 holds K = {8g..8g+7} and {16+8g..16+8g+7} (stride N).
        v16h bfrag[NT];
#pragma unroll
        for (int j = 0; j < NT; ++j) {
            const half_t* bp = Wt + (size_t)(k0 + g * 8) * N + nBlock + wn + j * 16 + m16;
#pragma unroll
            for (int s = 0; s < 8; ++s) {
                bfrag[j][s]     = bp[(size_t)s * N];
                bfrag[j][8 + s] = bp[(size_t)(16 + s) * N];
            }
        }
#pragma unroll
        for (int i = 0; i < 2; ++i)
#pragma unroll
            for (int j = 0; j < NT; ++j)
                acc[i][j] = __builtin_amdgcn_wmma_f32_16x16x32_f16(
                    false, afrag[i], false, bfrag[j],
                    (short)0, acc[i][j], false, false);
    }

    // ---- store: C/D layout: VGPR r -> M = r + 8*g, N = lane&15
#pragma unroll
    for (int i = 0; i < 2; ++i)
#pragma unroll
        for (int j = 0; j < NT; ++j)
#pragma unroll
            for (int r = 0; r < 8; ++r) {
                int row = mBlock + wm + i * 16 + r + g * 8;
                int col = nBlock + wn + j * 16 + m16;
                float v = acc[i][j][r];
                if (EPI == 1) v = softplusf(v + bias[col]);
                C[(size_t)row * N + col] = v;
            }
}

// =============================================================
// Weight transpose + f32 -> f16:  W[layers][N][K] -> Wt[layers][K][N]
// =============================================================
__global__ void w_transpose_f16_kernel(const float* __restrict__ W,
                                       half_t* __restrict__ Wt,
                                       int N, int K, long total)
{
    long gid = (long)blockIdx.x * blockDim.x + threadIdx.x;
    if (gid >= total) return;
    long nk    = (long)N * K;
    long layer = gid / nk;
    long rem   = gid % nk;
    int  n     = (int)(rem / K);
    int  k     = (int)(rem % K);
    Wt[layer * nk + (long)k * N + n] = (half_t)W[gid];
}

// =============================================================
// Embedding gather + LayerNorm -> hidden (f32).  One block per row.
// =============================================================
__global__ __launch_bounds__(256) void embed_ln_kernel(
    const int* __restrict__ tokens,
    const float* __restrict__ emb,
    const float* __restrict__ w,
    const float* __restrict__ b,
    float* __restrict__ hidden)
{
    __shared__ float ssum[256], ssq[256];
    const int row = blockIdx.x;
    const int t   = threadIdx.x;
    const int tok = tokens[row];
    const float* e = emb + (size_t)tok * D_MODEL;
    float v0 = e[t], v1 = e[t + 256];
    ssum[t] = v0 + v1;
    ssq[t]  = v0 * v0 + v1 * v1;
    __syncthreads();
    for (int off = 128; off > 0; off >>= 1) {
        if (t < off) { ssum[t] += ssum[t + off]; ssq[t] += ssq[t + off]; }
        __syncthreads();
    }
    float mean = ssum[0] * (1.f / D_MODEL);
    float var  = ssq[0] * (1.f / D_MODEL) - mean * mean;
    float rs   = rsqrtf(var + EPSLN);
    size_t base = (size_t)row * D_MODEL;
    hidden[base + t]       = (v0 - mean) * rs * w[t] + b[t];
    hidden[base + t + 256] = (v1 - mean) * rs * w[t + 256] + b[t + 256];
}

// =============================================================
// residual (+)= hidden ; hs16 = f16( LN(residual)*w + b )
// =============================================================
__global__ __launch_bounds__(256) void add_ln_pack_kernel(
    float* __restrict__ residual,
    const float* __restrict__ hidden,
    const float* __restrict__ w,
    const float* __restrict__ b,
    half_t* __restrict__ hs16,
    int first)
{
    __shared__ float ssum[256], ssq[256];
    const int row = blockIdx.x;
    const int t   = threadIdx.x;
    size_t base = (size_t)row * D_MODEL;
    float v0, v1;
    if (first) {
        v0 = hidden[base + t];
        v1 = hidden[base + t + 256];
    } else {
        v0 = residual[base + t] + hidden[base + t];
        v1 = residual[base + t + 256] + hidden[base + t + 256];
    }
    residual[base + t]       = v0;
    residual[base + t + 256] = v1;
    ssum[t] = v0 + v1;
    ssq[t]  = v0 * v0 + v1 * v1;
    __syncthreads();
    for (int off = 128; off > 0; off >>= 1) {
        if (t < off) { ssum[t] += ssum[t + off]; ssq[t] += ssq[t + off]; }
        __syncthreads();
    }
    float mean = ssum[0] * (1.f / D_MODEL);
    float var  = ssq[0] * (1.f / D_MODEL) - mean * mean;
    float rs   = rsqrtf(var + EPSLN);
    hs16[base + t]       = (half_t)((v0 - mean) * rs * w[t] + b[t]);
    hs16[base + t + 256] = (half_t)((v1 - mean) * rs * w[t + 256] + b[t + 256]);
}

// =============================================================
// Final: residual += hidden ; out = LN(residual) (f32 to d_out)
// =============================================================
__global__ __launch_bounds__(256) void final_ln_kernel(
    const float* __restrict__ residual,
    const float* __restrict__ hidden,
    const float* __restrict__ w,
    const float* __restrict__ b,
    float* __restrict__ out)
{
    __shared__ float ssum[256], ssq[256];
    const int row = blockIdx.x;
    const int t   = threadIdx.x;
    size_t base = (size_t)row * D_MODEL;
    float v0 = residual[base + t] + hidden[base + t];
    float v1 = residual[base + t + 256] + hidden[base + t + 256];
    ssum[t] = v0 + v1;
    ssq[t]  = v0 * v0 + v1 * v1;
    __syncthreads();
    for (int off = 128; off > 0; off >>= 1) {
        if (t < off) { ssum[t] += ssum[t + off]; ssq[t] += ssq[t + off]; }
        __syncthreads();
    }
    float mean = ssum[0] * (1.f / D_MODEL);
    float var  = ssq[0] * (1.f / D_MODEL) - mean * mean;
    float rs   = rsqrtf(var + EPSLN);
    out[base + t]       = (v0 - mean) * rs * w[t] + b[t];
    out[base + t + 256] = (v1 - mean) * rs * w[t + 256] + b[t + 256];
}

// =============================================================
// Causal depthwise conv (D_CONV=4) + SiLU.  x = xz[:, :, 0:1024].
// =============================================================
__global__ void conv_silu_kernel(const float* __restrict__ xz,
                                 const float* __restrict__ cw,  // [D_INNER][D_CONV]
                                 const float* __restrict__ cb,  // [D_INNER]
                                 half_t* __restrict__ xc16)
{
    long gid = (long)blockIdx.x * blockDim.x + threadIdx.x;
    if (gid >= (long)MTOT * D_INNER) return;
    int d  = (int)(gid % D_INNER);
    long bl = gid / D_INNER;
    int l  = (int)(bl % SEQ);
    long b = bl / SEQ;
    float acc = cb[d];
#pragma unroll
    for (int k = 0; k < D_CONV; ++k) {
        int ls = l + k - (D_CONV - 1);
        if (ls >= 0)
            acc += xz[((size_t)(b * SEQ + ls)) * (2 * D_INNER) + d] * cw[d * D_CONV + k];
    }
    xc16[gid] = (half_t)siluf(acc);
}

// =============================================================
// dtr16 = f16(x_dbl[:, 0:32])
// =============================================================
__global__ void split_dtr_kernel(const float* __restrict__ xdbl,
                                 half_t* __restrict__ dtr16)
{
    long gid = (long)blockIdx.x * blockDim.x + threadIdx.x;
    if (gid >= (long)MTOT * DT_RANK) return;
    long row = gid / DT_RANK;
    int  c   = (int)(gid % DT_RANK);
    dtr16[gid] = (half_t)xdbl[row * (DT_RANK + 2 * D_STATE) + c];
}

// =============================================================
// Selective scan: one thread per (batch, channel). Sequential in L.
// B/C taken in-place from x_dbl cols [32:48) / [48:64) as float4s.
// =============================================================
__global__ __launch_bounds__(256) void scan_kernel(
    const float* __restrict__ dt,      // [MTOT][D_INNER]
    const half_t* __restrict__ x16,    // [MTOT][D_INNER]
    const float* __restrict__ xdbl,    // [MTOT][64]
    const float* __restrict__ A_log,   // [D_INNER][D_STATE] (layer slice)
    float* __restrict__ y)             // [MTOT][D_INNER]
{
    int gid = blockIdx.x * blockDim.x + threadIdx.x;
    if (gid >= BATCH * D_INNER) return;
    int b = gid / D_INNER;
    int d = gid % D_INNER;

    float Arow[D_STATE], h[D_STATE];
#pragma unroll
    for (int s = 0; s < D_STATE; ++s) {
        Arow[s] = -__expf(A_log[d * D_STATE + s]);
        h[s] = 0.f;
    }
    for (int l = 0; l < SEQ; ++l) {
        size_t idx = (size_t)b * SEQ + l;
        float dtv = dt[idx * D_INNER + d];
        float xv  = (float)x16[idx * D_INNER + d];
        float dx  = dtv * xv;
        const v4f* Bv = (const v4f*)(xdbl + idx * 64 + DT_RANK);
        const v4f* Cv = (const v4f*)(xdbl + idx * 64 + DT_RANK + D_STATE);
        float acc = 0.f;
#pragma unroll
        for (int q = 0; q < 4; ++q) {
            v4f Bq = Bv[q];
            v4f Cq = Cv[q];
#pragma unroll
            for (int u = 0; u < 4; ++u) {
                int s = q * 4 + u;
                float dA = __expf(dtv * Arow[s]);
                h[s] = dA * h[s] + dx * Bq[u];
                acc += h[s] * Cq[u];
            }
        }
        y[idx * D_INNER + d] = acc;
    }
}

// =============================================================
// yg16 = f16( (y + x*Dp) * silu(z) ),  z = xz[:, :, 1024:2048]
// =============================================================
__global__ void gate_kernel(const float* __restrict__ y,
                            const half_t* __restrict__ x16,
                            const float* __restrict__ xz,
                            const float* __restrict__ Dp,
                            half_t* __restrict__ yg16)
{
    long gid = (long)blockIdx.x * blockDim.x + threadIdx.x;
    if (gid >= (long)MTOT * D_INNER) return;
    int d  = (int)(gid % D_INNER);
    long bl = gid / D_INNER;
    float yv = y[gid] + (float)x16[gid] * Dp[d];
    float zv = xz[(size_t)bl * (2 * D_INNER) + D_INNER + d];
    yg16[gid] = (half_t)(yv * siluf(zv));
}

// =============================================================
// Launch
// =============================================================
extern "C" void kernel_launch(void* const* d_in, const int* in_sizes, int n_in,
                              void* d_out, int out_size, void* d_ws, size_t ws_size,
                              hipStream_t stream) {
    const int*   tokens    = (const int*)  d_in[0];
    // d_in[1] = pad_mask (unused by reference math)
    const float* emb_table = (const float*)d_in[2];
    const float* enc_ln_w  = (const float*)d_in[3];
    const float* enc_ln_b  = (const float*)d_in[4];
    const float* in_proj_w = (const float*)d_in[5];   // [4][2048][512]
    const float* conv_w    = (const float*)d_in[6];   // [4][1024][4]
    const float* conv_b    = (const float*)d_in[7];   // [4][1024]
    const float* xproj_w   = (const float*)d_in[8];   // [4][64][1024]
    const float* dtproj_w  = (const float*)d_in[9];   // [4][1024][32]
    const float* dtproj_b  = (const float*)d_in[10];  // [4][1024]
    const float* A_log     = (const float*)d_in[11];  // [4][1024][16]
    const float* D_param   = (const float*)d_in[12];  // [4][1024]
    const float* out_proj_w= (const float*)d_in[13];  // [4][512][1024]
    const float* ln_w      = (const float*)d_in[14];  // [4][512]
    const float* ln_b      = (const float*)d_in[15];  // [4][512]
    const float* normf_w   = (const float*)d_in[16];
    const float* normf_b   = (const float*)d_in[17];
    float* out = (float*)d_out;

    // ---------- workspace carve-out ----------
    unsigned char* p = (unsigned char*)d_ws;
    auto carve = [&](size_t bytes) -> void* {
        void* r = (void*)p;
        p += (bytes + 255) & ~(size_t)255;
        return r;
    };
    half_t* inprojT  = (half_t*)carve((size_t)NLAYERS * D_MODEL * 2 * D_INNER * sizeof(half_t)); // [4][512][2048]
    half_t* xprojT   = (half_t*)carve((size_t)NLAYERS * D_INNER * 64 * sizeof(half_t));          // [4][1024][64]
    half_t* dtprojT  = (half_t*)carve((size_t)NLAYERS * DT_RANK * D_INNER * sizeof(half_t));     // [4][32][1024]
    half_t* outprojT = (half_t*)carve((size_t)NLAYERS * D_INNER * D_MODEL * sizeof(half_t));     // [4][1024][512]
    float*  residual = (float*) carve((size_t)MTOT * D_MODEL * sizeof(float));
    float*  hidden   = (float*) carve((size_t)MTOT * D_MODEL * sizeof(float));
    half_t* hs16     = (half_t*)carve((size_t)MTOT * D_MODEL * sizeof(half_t));
    float*  xz       = (float*) carve((size_t)MTOT * 2 * D_INNER * sizeof(float));
    half_t* xc16     = (half_t*)carve((size_t)MTOT * D_INNER * sizeof(half_t));
    float*  xdbl     = (float*) carve((size_t)MTOT * 64 * sizeof(float));
    half_t* dtr16    = (half_t*)carve((size_t)MTOT * DT_RANK * sizeof(half_t));
    float*  dtbuf    = (float*) carve((size_t)MTOT * D_INNER * sizeof(float));
    float*  ybuf     = (float*) carve((size_t)MTOT * D_INNER * sizeof(float));
    half_t* yg16     = (half_t*)carve((size_t)MTOT * D_INNER * sizeof(half_t));
    (void)ws_size; (void)in_sizes; (void)n_in; (void)out_size;

    // ---------- weight transpose/convert (all layers at once) ----------
    {
        long t1 = (long)NLAYERS * (2 * D_INNER) * D_MODEL;
        w_transpose_f16_kernel<<<(t1 + 255) / 256, 256, 0, stream>>>(in_proj_w, inprojT, 2 * D_INNER, D_MODEL, t1);
        long t2 = (long)NLAYERS * 64 * D_INNER;
        w_transpose_f16_kernel<<<(t2 + 255) / 256, 256, 0, stream>>>(xproj_w, xprojT, 64, D_INNER, t2);
        long t3 = (long)NLAYERS * D_INNER * DT_RANK;
        w_transpose_f16_kernel<<<(t3 + 255) / 256, 256, 0, stream>>>(dtproj_w, dtprojT, D_INNER, DT_RANK, t3);
        long t4 = (long)NLAYERS * D_MODEL * D_INNER;
        w_transpose_f16_kernel<<<(t4 + 255) / 256, 256, 0, stream>>>(out_proj_w, outprojT, D_MODEL, D_INNER, t4);
    }

    // ---------- embedding + encoder LN ----------
    embed_ln_kernel<<<MTOT, 256, 0, stream>>>(tokens, emb_table, enc_ln_w, enc_ln_b, hidden);

    const long ebig = (long)MTOT * D_INNER;   // 16.7M elementwise threads
    for (int i = 0; i < NLAYERS; ++i) {
        // residual += hidden ; hs16 = LN(residual)
        add_ln_pack_kernel<<<MTOT, 256, 0, stream>>>(
            residual, hidden, ln_w + i * D_MODEL, ln_b + i * D_MODEL, hs16, (i == 0) ? 1 : 0);

        // GEMM1: xz = hs16 (M x 512) * inprojT (512 x 2048)   [block 128x128]
        gemm_wmma_kernel<0, 4><<<dim3((2 * D_INNER) / 128, MTOT / 128), 256, 0, stream>>>(
            hs16, inprojT + (size_t)i * D_MODEL * 2 * D_INNER, xz, nullptr,
            MTOT, 2 * D_INNER, D_MODEL);

        // causal conv + SiLU -> xc16
        conv_silu_kernel<<<(ebig + 255) / 256, 256, 0, stream>>>(
            xz, conv_w + (size_t)i * D_INNER * D_CONV, conv_b + (size_t)i * D_INNER, xc16);

        // GEMM2: x_dbl = xc16 (M x 1024) * xprojT (1024 x 64)  [block 128x64]
        gemm_wmma_kernel<0, 2><<<dim3(64 / 64, MTOT / 128), 256, 0, stream>>>(
            xc16, xprojT + (size_t)i * D_INNER * 64, xdbl, nullptr,
            MTOT, 64, D_INNER);

        // dtr16 = f16(x_dbl[:, :32])
        long tspl = (long)MTOT * DT_RANK;
        split_dtr_kernel<<<(tspl + 255) / 256, 256, 0, stream>>>(xdbl, dtr16);

        // GEMM3 (+bias +softplus): dt = softplus(dtr16 (M x 32) * dtprojT (32 x 1024) + dtb)
        gemm_wmma_kernel<1, 4><<<dim3(D_INNER / 128, MTOT / 128), 256, 0, stream>>>(
            dtr16, dtprojT + (size_t)i * DT_RANK * D_INNER, dtbuf,
            dtproj_b + (size_t)i * D_INNER, MTOT, D_INNER, DT_RANK);

        // selective scan
        scan_kernel<<<(BATCH * D_INNER) / 256, 256, 0, stream>>>(
            dtbuf, xc16, xdbl, A_log + (size_t)i * D_INNER * D_STATE, ybuf);

        // gating: yg16 = (y + x*Dp) * silu(z)
        gate_kernel<<<(ebig + 255) / 256, 256, 0, stream>>>(
            ybuf, xc16, xz, D_param + (size_t)i * D_INNER, yg16);

        // GEMM4: hidden = yg16 (M x 1024) * outprojT (1024 x 512)  [block 128x128]
        gemm_wmma_kernel<0, 4><<<dim3(D_MODEL / 128, MTOT / 128), 256, 0, stream>>>(
            yg16, outprojT + (size_t)i * D_INNER * D_MODEL, hidden, nullptr,
            MTOT, D_MODEL, D_INNER);
    }

    // ---------- final residual add + LN -> d_out ----------
    final_ln_kernel<<<MTOT, 256, 0, stream>>>(residual, hidden, normf_w, normf_b, out);
}
